// InvRadonFourierLayer_70566312673314
// MI455X (gfx1250) — matrix operand order repaired and checked by reading, not compiled, not run on hardware
//
#include <hip/hip_runtime.h>
#include <math.h>

// Problem constants (fixed by setup_inputs): N=8, H=768, W=360, D=512
#define NIMG 8
#define HDIM 768
#define WDIM 360
#define DDIM 512

typedef __attribute__((ext_vector_type(16))) _Float16 v16h;
typedef __attribute__((ext_vector_type(8)))  float    v8f;

// ---------------------------------------------------------------------------
// Kernel 1: spatial ramp kernel  k[j] = (1/H) * sum_m h[m] * cos(2*pi*m*j/H)
// Exact integer phase reduction: (m*j) mod H accumulated incrementally.
// ---------------------------------------------------------------------------
__global__ void ramp_spatial_kernel(const float* __restrict__ hG,
                                    float* __restrict__ kOut) {
    int j = blockIdx.x * blockDim.x + threadIdx.x;
    if (j >= HDIM) return;
    const float twopi_over_H = 6.28318530717958647692f / (float)HDIM;
    float acc = 0.0f;
    int r = 0; // (m*j) mod H
    for (int m = 0; m < HDIM; ++m) {
        acc += hG[m] * __cosf((float)r * twopi_over_H);
        r += j;
        if (r >= HDIM) r -= HDIM;
    }
    kOut[j] = acc * (1.0f / (float)HDIM);
}

// ---------------------------------------------------------------------------
// Kernel 2: circulant-convolution GEMM via WMMA (split-f16, f32 accumulate)
//   filt[n,i,w] = sum_j k[(i-j) mod H] * radon[n,j,w]
// A (M=768 x K=768) circulant, B (K=768 x Ncols=2880), cols = n*W + w.
// One wave per 16x16 output tile; 8 waves (256 threads) per block.
// A fragment layout (16-bit A 16x32, ISA 7.12.2):
//   lane g=lane>>4, m=lane&15; element e=2*vv+hb ->
//   Koff = (vv<4 ? 2*vv+hb : 16+2*(vv-4)+hb) + 8*g
// B fragment (32x16, K-striped): lane n=lane&15; element e -> K = (e<8 ? 8*g+e : 16+8*g+(e-8)), same as A
// C/D: VGPR r -> (M = r + 8*g, N = lane&15)
// ---------------------------------------------------------------------------
__global__ void __launch_bounds__(256)
filter_gemm_wmma(const float* __restrict__ radon,
                 const float* __restrict__ kG,
                 float* __restrict__ filt) {
    __shared__ float kLDS[HDIM];
    const int tid = threadIdx.x;
    for (int idx = tid; idx < HDIM; idx += 256) kLDS[idx] = kG[idx];
    __syncthreads();

    const int lane = tid & 31;
    const int wave = tid >> 5;
    const int tileId = blockIdx.x * 8 + wave;   // 0 .. 8639
    const int mt = tileId % 48;                  // 48 M-tiles  (768/16)
    const int nt = tileId / 48;                  // 180 N-tiles (2880/16)

    const int g  = lane >> 4;   // lane half
    const int ml = lane & 15;

    const int i_row = mt * 16 + ml;      // A row (M) for this lane
    const int col   = nt * 16 + ml;      // output column for this lane
    const int n_img = col / WDIM;
    const int wcol  = col - n_img * WDIM;
    const float* __restrict__ Bcol = radon + (size_t)n_img * HDIM * WDIM + wcol;

    v8f c = {};
    for (int kbase = 0; kbase < HDIM; kbase += 32) {
        v16h aHi, aLo, bHi, bLo;
#pragma unroll
        for (int e = 0; e < 16; ++e) {
            const int vv = e >> 1, hb = e & 1;
            const int Koff = ((vv < 4) ? (2 * vv + hb)
                                       : (16 + 2 * (vv - 4) + hb)) + 8 * g;
            int d = i_row - (kbase + Koff);
            if (d < 0) d += HDIM;
            const float a = kLDS[d];
            const _Float16 ah = (_Float16)a;
            aHi[e] = ah;
            aLo[e] = (_Float16)(a - (float)ah);
        }
#pragma unroll
        for (int e = 0; e < 16; ++e) {
            const int K = kbase + ((e < 8) ? (8 * g + e) : (16 + 8 * g + (e - 8)));
            const float b = Bcol[(size_t)K * WDIM];
            const _Float16 bh = (_Float16)b;
            bHi[e] = bh;
            bLo[e] = (_Float16)(b - (float)bh);
        }
        // D = A*B + C with split-precision correction terms
        c = __builtin_amdgcn_wmma_f32_16x16x32_f16(false, aHi, false, bHi,
                                                   (short)0, c, false, false);
        c = __builtin_amdgcn_wmma_f32_16x16x32_f16(false, aHi, false, bLo,
                                                   (short)0, c, false, false);
        c = __builtin_amdgcn_wmma_f32_16x16x32_f16(false, aLo, false, bHi,
                                                   (short)0, c, false, false);
    }

    float* __restrict__ Dcol = filt + (size_t)n_img * HDIM * WDIM + wcol;
#pragma unroll
    for (int r = 0; r < 8; ++r) {
        const int i_out = mt * 16 + r + 8 * g;
        Dcol[(size_t)i_out * WDIM] = c[r];
    }
}

// ---------------------------------------------------------------------------
// Kernel 3: backprojection. 16x16 pixel tile per 256-thread block.
// Faithful to reference: px = (tx+1)*sx (nearly integer), bilinear in y,
// bounds -> zero, final scale pi/(2W).
// ---------------------------------------------------------------------------
__global__ void __launch_bounds__(256)
backproject_kernel(const float* __restrict__ filt,
                   float* __restrict__ out) {
    __shared__ float cosS[WDIM];
    __shared__ float sinS[WDIM];
    __shared__ int   x0S[WDIM];
    __shared__ float wxS[WDIM];

    const int tid = threadIdx.x;
    for (int a = tid; a < WDIM; a += 256) {
        const float th = (float)a * (float)(M_PI / 360.0);   // pi/180 * 0.5*a
        cosS[a] = __cosf(th);
        sinS[a] = __sinf(th);
        const float tx = -1.0f + (float)a * (2.0f / (float)(WDIM - 1));
        const float px = (tx + 1.0f) * ((float)(WDIM - 1) * 0.5f);
        const float f  = floorf(px);
        x0S[a] = (int)f;
        wxS[a] = px - f;
    }
    __syncthreads();

    const int bid  = blockIdx.x;
    const int n    = bid >> 10;           // 1024 tiles per image
    const int tile = bid & 1023;
    const int ti   = tile >> 5;
    const int tj   = tile & 31;
    const int i = ti * 16 + (tid >> 4);   // Xg index
    const int j = tj * 16 + (tid & 15);   // Yg index

    const float xpr = (float)(i - DDIM / 2);
    const float ypr = (float)(j - DDIM / 2);
    const float sy  = (float)(HDIM - 1) * 0.5f;   // 383.5
    const float inv_hH = 1.0f / (float)(HDIM / 2);

    const float* __restrict__ fN = filt + (size_t)n * HDIM * WDIM;

    float acc = 0.0f;
    for (int a = 0; a < WDIM; ++a) {
        const float t  = ypr * cosS[a] - xpr * sinS[a];
        const float py = (t * inv_hH + 1.0f) * sy;
        const float fy = floorf(py);
        const int   y0 = (int)fy;
        const float wy = py - fy;
        const int   x0 = x0S[a];
        const float wx = wxS[a];

        // sample column xi with y-bilinear + bounds-to-zero
        auto samp = [&](int xi) -> float {
            if (xi < 0 || xi >= WDIM) return 0.0f;
            const float v0 = (y0 >= 0 && y0 < HDIM)
                                 ? fN[(size_t)y0 * WDIM + xi] : 0.0f;
            const int y1 = y0 + 1;
            const float v1 = (y1 >= 0 && y1 < HDIM)
                                 ? fN[(size_t)y1 * WDIM + xi] : 0.0f;
            return v0 * (1.0f - wy) + v1 * wy;
        };
        acc += samp(x0) * (1.0f - wx) + samp(x0 + 1) * wx;
    }

    out[((size_t)n * DDIM + i) * DDIM + j] = acc * (float)(M_PI / (2.0 * WDIM));
}

// ---------------------------------------------------------------------------
extern "C" void kernel_launch(void* const* d_in, const int* in_sizes, int n_in,
                              void* d_out, int out_size, void* d_ws, size_t ws_size,
                              hipStream_t stream) {
    const float* radon = (const float*)d_in[0];   // (8,1,768,360)
    const float* hG    = (const float*)d_in[1];   // 768 floats
    float* out = (float*)d_out;                   // (8,1,512,512)

    float* wsf   = (float*)d_ws;
    float* kSpat = wsf;                            // 768 floats
    float* filt  = wsf + 1024;                     // 8*768*360 floats (~8.85 MB)

    // 1) spatial ramp kernel from hG
    ramp_spatial_kernel<<<3, 256, 0, stream>>>(hG, kSpat);

    // 2) circulant convolution as WMMA GEMM: 8640 tiles, 8 waves/block
    filter_gemm_wmma<<<1080, 256, 0, stream>>>(radon, kSpat, filt);

    // 3) backprojection: 8 images * 32*32 tiles of 16x16 pixels
    backproject_kernel<<<NIMG * 1024, 256, 0, stream>>>(filt, out);
}
